// theEncoderblock_56032143344083
// MI455X (gfx1250) — matrix operand order repaired and checked by reading, not compiled, not run on hardware
//
#include <hip/hip_runtime.h>
#include <hip/hip_bf16.h>
#include <cstdint>
#include <cstddef>

// Problem constants (match reference)
#define BB   16
#define CC   512
#define LL   1024
#define KW   7
#define HH   8
#define DHH  64
#define ATTN_WPB 8   // waves per attention block

typedef __attribute__((ext_vector_type(16))) __bf16 bf16x16;
typedef __attribute__((ext_vector_type(8)))  float  floatx8;

union ABFrag {
    uint4          u4[2];
    unsigned short s[16];
    bf16x16        v;
};

__device__ __forceinline__ unsigned short f2bf(float f) {
    union { float f; unsigned u; } cv; cv.f = f;
    unsigned r = cv.u + 0x7FFFu + ((cv.u >> 16) & 1u);   // round-to-nearest-even
    return (unsigned short)(r >> 16);
}

__device__ __forceinline__ floatx8 wmma_bf16(const bf16x16& a, const bf16x16& b, const floatx8& c) {
    return __builtin_amdgcn_wmma_f32_16x16x32_bf16(false, a, false, b, (short)0, c, false, false);
}

// ---------------------------------------------------------------------------
// 1) Positional encoding: out[b,c,l] = x[b,c,l] + (c<256 ? sin : cos)(l*inv)
// ---------------------------------------------------------------------------
__global__ __launch_bounds__(256) void k_posenc(const float* __restrict__ x, float* __restrict__ out) {
    int idx = blockIdx.x * 256 + threadIdx.x;
    int l = idx % LL;
    int c = (idx / LL) % CC;
    const float log_inc = __logf(10000.0f) / 255.0f;
    int j = (c < 256) ? c : (c - 256);
    float inv = __expf(-(float)j * log_inc);
    float ph  = (float)l * inv;
    float sig = (c < 256) ? __sinf(ph) : __cosf(ph);
    out[idx] = x[idx] + sig;
}

// ---------------------------------------------------------------------------
// 2) Channel LayerNorm (normalize over C at each (b,l)); coalesced over l.
// ---------------------------------------------------------------------------
__global__ __launch_bounds__(256) void k_ln_ch(const float* __restrict__ x,
                                               const float* __restrict__ sc,
                                               const float* __restrict__ bi,
                                               float* __restrict__ out) {
    int l = blockIdx.x * 256 + threadIdx.x;
    int b = blockIdx.y;
    size_t base = (size_t)b * CC * LL + l;
    float sum = 0.f, sq = 0.f;
    for (int c = 0; c < CC; ++c) {
        float v = x[base + (size_t)c * LL];
        sum += v; sq += v * v;
    }
    float mu  = sum * (1.0f / CC);
    float var = sq * (1.0f / CC) - mu * mu;
    float r   = rsqrtf(var + 1e-5f);
    for (int c = 0; c < CC; ++c) {
        size_t i = base + (size_t)c * LL;
        out[i] = (x[i] - mu) * r * sc[c] + bi[c];
    }
}

// ---------------------------------------------------------------------------
// 3) Depthwise conv K=7, same padding
// ---------------------------------------------------------------------------
__global__ __launch_bounds__(256) void k_dwconv(const float* __restrict__ x,
                                                const float* __restrict__ w,
                                                float* __restrict__ out) {
    int idx = blockIdx.x * 256 + threadIdx.x;
    int l = idx % LL;
    int c = (idx / LL) % CC;
    size_t rowbase = (size_t)idx - l;
    float acc = 0.f;
#pragma unroll
    for (int k = 0; k < KW; ++k) {
        int ll = l + k - (KW / 2);
        if (ll >= 0 && ll < LL) acc += x[rowbase + ll] * w[c * KW + k];
    }
    out[idx] = acc;
}

// ---------------------------------------------------------------------------
// 4) f32 -> bf16 bit-pattern copy (weights)
// ---------------------------------------------------------------------------
__global__ __launch_bounds__(256) void k_f2bf(const float* __restrict__ src,
                                              unsigned short* __restrict__ dst, int n) {
    int idx = blockIdx.x * 256 + threadIdx.x;
    if (idx < n) dst[idx] = f2bf(src[idx]);
}

// ---------------------------------------------------------------------------
// 5) Transpose+convert: f32 [B,C,L] -> bf16 [B,L,C]  (K-contiguous B operand)
// ---------------------------------------------------------------------------
__global__ __launch_bounds__(256) void k_transpose(const float* __restrict__ src,
                                                   unsigned short* __restrict__ dst) {
    __shared__ float t[32][33];
    int b  = blockIdx.z;
    int l0 = blockIdx.x * 32;
    int c0 = blockIdx.y * 32;
    int tx = threadIdx.x, ty = threadIdx.y;
#pragma unroll
    for (int j = ty; j < 32; j += 8)
        t[j][tx] = src[((size_t)b * CC + (c0 + j)) * LL + l0 + tx];
    __syncthreads();
#pragma unroll
    for (int j = ty; j < 32; j += 8)
        dst[((size_t)b * LL + (l0 + j)) * CC + c0 + tx] = f2bf(t[tx][j]);
}

// ---------------------------------------------------------------------------
// 6) WMMA GEMM (software-pipelined): out[b,m,l] = act(W.X + bias) (+ resid)
//    Wb: [Mtot,512] bf16 row-major;  Xt: [B,L,512] bf16 (K-contiguous per l).
//    One wave -> 16(M) x 64(N) tile; K=512 in 16 WMMA K=32 steps.
//    Double-buffered A/B fragments so VMEM overlaps the matrix pipe.
// ---------------------------------------------------------------------------
__global__ __launch_bounds__(256) void k_gemm(const unsigned short* __restrict__ Wb,
                                              const unsigned short* __restrict__ Xt,
                                              const float* __restrict__ bias,
                                              const float* resid,
                                              float* out,
                                              int Mtot, int doRelu) {
    const int KD = CC;
    int wave = (blockIdx.x * blockDim.x + threadIdx.x) >> 5;
    int lane = threadIdx.x & 31;
    int mtiles  = Mtot >> 4;
    int ngroups = LL >> 6;                 // 16
    int nwaves  = BB * mtiles * ngroups;
    if (wave >= nwaves) return;
    int ng = wave % ngroups;
    int t2 = wave / ngroups;
    int mt = t2 % mtiles;
    int b  = t2 / mtiles;

    int m0   = mt << 4;
    int n0   = ng << 6;
    int arow = m0 + (lane & 15);
    int kA   = (lane < 16) ? 0 : 16;       // A-frag K start (contiguous K order)
    int kBo  = (lane < 16) ? 0 : 16;       // B-frag K start
    const unsigned short* Arow  = Wb + (size_t)arow * KD;
    const unsigned short* Bcol0 = Xt + (size_t)b * LL * KD
                                     + (size_t)(n0 + (lane & 15)) * KD + kBo;

    floatx8 acc[4] = {};
    ABFrag a[2], bb[2][4];

    // prologue: load phase 0
    a[0].u4[0] = *(const uint4*)(Arow + kA);
    a[0].u4[1] = *(const uint4*)(Arow + kA + 8);
#pragma unroll
    for (int j = 0; j < 4; ++j) {
        const unsigned short* p = Bcol0 + (size_t)(j << 4) * KD;
        bb[0][j].u4[0] = *(const uint4*)(p);
        bb[0][j].u4[1] = *(const uint4*)(p + 8);
    }

    for (int k0 = 0; k0 < KD; k0 += 32) {
        int ph = (k0 >> 5) & 1;
        int np = ph ^ 1;
        int nxt = k0 + 32;
        if (nxt < KD) {
            a[np].u4[0] = *(const uint4*)(Arow + nxt + kA);
            a[np].u4[1] = *(const uint4*)(Arow + nxt + kA + 8);
            __builtin_prefetch(Arow + nxt + 32 + kA, 0, 1);
#pragma unroll
            for (int j = 0; j < 4; ++j) {
                const unsigned short* p = Bcol0 + (size_t)(j << 4) * KD + nxt;
                bb[np][j].u4[0] = *(const uint4*)(p);
                bb[np][j].u4[1] = *(const uint4*)(p + 8);
            }
        }
#pragma unroll
        for (int j = 0; j < 4; ++j)
            acc[j] = wmma_bf16(a[ph].v, bb[ph][j].v, acc[j]);
    }
#pragma unroll
    for (int j = 0; j < 4; ++j) {
        int col = n0 + (j << 4) + (lane & 15);
#pragma unroll
        for (int v = 0; v < 8; ++v) {
            int r = m0 + v + ((lane >> 4) << 3);
            float val = acc[j][v];
            if (bias) val += bias[r];
            if (doRelu) val = val > 0.f ? val : 0.f;
            size_t oi = ((size_t)b * Mtot + r) * LL + col;
            if (resid) val += resid[oi];
            out[oi] = val;
        }
    }
}

// ---------------------------------------------------------------------------
// 7) Pack Q/K/V into WMMA-friendly bf16 layouts.
//    Qb[b,h,lq,dh] = Q[b,h*64+dh,lq] * DH^-0.5      (A operand, dh contiguous)
//    Kb[b,h,lk,dh] = KV[b,h*64+dh,lk]               (B operand of QK^T)
//    Vb[b,h,dh,lk] = KV[b,C+h*64+dh,lk]             (B operand of P.V)
// ---------------------------------------------------------------------------
__global__ __launch_bounds__(256) void k_pack(const float* __restrict__ Qf,
                                              const float* __restrict__ KVf,
                                              unsigned short* __restrict__ Qb,
                                              unsigned short* __restrict__ Kb,
                                              unsigned short* __restrict__ Vb) {
    int idx = blockIdx.x * 256 + threadIdx.x;
    int l = idx % LL;
    int c = (idx / LL) % CC;
    int b = idx / (LL * CC);
    int h  = c >> 6;
    int dh = c & 63;
    size_t bh = (size_t)(b * HH + h);
    Qb[(bh * LL + l) * DHH + dh] = f2bf(Qf[idx] * 0.125f);   // 64^-0.5
    Kb[(bh * LL + l) * DHH + dh] = f2bf(KVf[((size_t)b * 2 * CC + c) * LL + l]);
    Vb[(bh * DHH + dh) * LL + l] = f2bf(KVf[((size_t)b * 2 * CC + CC + c) * LL + l]);
}

// ---------------------------------------------------------------------------
// 8) Flash-style attention: per wave = one (b,h,16-row q-tile).
//    Loop over 32-wide k-tiles: S=QK^T (4 WMMAs), online softmax with
//    half-wave shuffles, P transposed through a PRIVATE per-wave LDS
//    partition (same-wave LDS ops are processed in order per the ISA, so
//    only a compile-time fence is needed -- no s_barrier), O += P.V (4 WMMAs).
// ---------------------------------------------------------------------------
__global__ __launch_bounds__(32 * ATTN_WPB) void k_attn(const unsigned short* __restrict__ Qb,
                                                        const unsigned short* __restrict__ Kb,
                                                        const unsigned short* __restrict__ Vb,
                                                        const int* __restrict__ mask,
                                                        const float* resid,
                                                        float* out) {
    __shared__ unsigned short plds[ATTN_WPB][16][36];
    int lane = threadIdx.x & 31;
    int wv   = threadIdx.x >> 5;
    int gw   = blockIdx.x * ATTN_WPB + wv;
    int qt   = gw % (LL / 16);
    int t2   = gw / (LL / 16);
    int h    = t2 % HH;
    int b    = t2 / HH;
    size_t bh = (size_t)(b * HH + h);

    int nl   = lane & 15;
    int half = lane >> 4;
    int kA   = (lane < 16) ? 0 : 16;
    int kBo  = (lane < 16) ? 0 : 16;

    // Q A-fragments for K(dh)=0..31 and 32..63
    const unsigned short* qp = Qb + (bh * LL + (size_t)qt * 16 + nl) * DHH;
    ABFrag aq0, aq1;
    aq0.u4[0] = *(const uint4*)(qp + kA);
    aq0.u4[1] = *(const uint4*)(qp + kA + 8);
    aq1.u4[0] = *(const uint4*)(qp + 32 + kA);
    aq1.u4[1] = *(const uint4*)(qp + 32 + kA + 8);

    float mrow[8], ssum[8];
#pragma unroll
    for (int v = 0; v < 8; ++v) { mrow[v] = -3.0e38f; ssum[v] = 0.f; }
    floatx8 o[4] = {};

    const unsigned short* kbase = Kb + bh * LL * DHH;
    const unsigned short* vbase = Vb + bh * DHH * LL;
    const int* mb = mask + b * LL;

    for (int kt = 0; kt < LL; kt += 32) {
        floatx8 s0 = {}, s1 = {};
        {   // n-tile 0: k columns kt..kt+15
            const unsigned short* p = kbase + (size_t)(kt + nl) * DHH;
            ABFrag bk0, bk1;
            bk0.u4[0] = *(const uint4*)(p + kBo);
            bk0.u4[1] = *(const uint4*)(p + kBo + 8);
            bk1.u4[0] = *(const uint4*)(p + 32 + kBo);
            bk1.u4[1] = *(const uint4*)(p + 32 + kBo + 8);
            s0 = wmma_bf16(aq0.v, bk0.v, s0);
            s0 = wmma_bf16(aq1.v, bk1.v, s0);
        }
        {   // n-tile 1: k columns kt+16..kt+31
            const unsigned short* p = kbase + (size_t)(kt + 16 + nl) * DHH;
            ABFrag bk0, bk1;
            bk0.u4[0] = *(const uint4*)(p + kBo);
            bk0.u4[1] = *(const uint4*)(p + kBo + 8);
            bk1.u4[0] = *(const uint4*)(p + 32 + kBo);
            bk1.u4[1] = *(const uint4*)(p + 32 + kBo + 8);
            s1 = wmma_bf16(aq0.v, bk0.v, s1);
            s1 = wmma_bf16(aq1.v, bk1.v, s1);
        }
        // mask (lane owns one k column per fragment)
        float madd0 = mb[kt + nl]      ? 0.f : -1.0e30f;
        float madd1 = mb[kt + 16 + nl] ? 0.f : -1.0e30f;
        float p0[8], p1[8];
#pragma unroll
        for (int v = 0; v < 8; ++v) { s0[v] += madd0; s1[v] += madd1; }
        // online softmax: row reductions within each 16-lane half-wave
#pragma unroll
        for (int v = 0; v < 8; ++v) {
            float m = fmaxf(s0[v], s1[v]);
            for (int d = 1; d < 16; d <<= 1) m = fmaxf(m, __shfl_xor(m, d, 16));
            float mn = fmaxf(mrow[v], m);
            float sc = __expf(mrow[v] - mn);
            mrow[v]  = mn;
            p0[v] = __expf(s0[v] - mn);
            p1[v] = __expf(s1[v] - mn);
            float rs = p0[v] + p1[v];
            for (int d = 1; d < 16; d <<= 1) rs += __shfl_xor(rs, d, 16);
            ssum[v] = ssum[v] * sc + rs;
#pragma unroll
            for (int j = 0; j < 4; ++j) o[j][v] *= sc;
        }
        // transpose P (C-frag layout) -> A-frag layout via private LDS slice.
        // Same-wave LDS ops complete in order (DScnt in-order per ISA); only a
        // compile-time fence is required to keep program order.
#pragma unroll
        for (int v = 0; v < 8; ++v) {
            int r = v + (half << 3);
            plds[wv][r][nl]      = f2bf(p0[v]);
            plds[wv][r][16 + nl] = f2bf(p1[v]);
        }
        asm volatile("" ::: "memory");
        ABFrag pa;
#pragma unroll
        for (int e = 0; e < 8; ++e) {
            pa.s[e]     = plds[wv][nl][kA + e];
            pa.s[8 + e] = plds[wv][nl][kA + 8 + e];
        }
        asm volatile("" ::: "memory");
        // O += P . V
#pragma unroll
        for (int j = 0; j < 4; ++j) {
            int ncol = (j << 4) + nl;               // dh column
            const unsigned short* p = vbase + (size_t)ncol * LL + kt + kBo;
            ABFrag bv;
            bv.u4[0] = *(const uint4*)(p);
            bv.u4[1] = *(const uint4*)(p + 8);
            o[j] = wmma_bf16(pa.v, bv.v, o[j]);
        }
    }
    // epilogue: out = O / rowsum + residual
#pragma unroll
    for (int j = 0; j < 4; ++j) {
        int dh = (j << 4) + nl;
        int c  = h * DHH + dh;
#pragma unroll
        for (int v = 0; v < 8; ++v) {
            int lq = qt * 16 + v + (half << 3);
            size_t idx = ((size_t)b * CC + c) * LL + lq;
            out[idx] = o[j][v] / ssum[v] + resid[idx];
        }
    }
}

// ---------------------------------------------------------------------------
// Host launcher
// ---------------------------------------------------------------------------
extern "C" void kernel_launch(void* const* d_in, const int* in_sizes, int n_in,
                              void* d_out, int out_size, void* d_ws, size_t ws_size,
                              hipStream_t stream) {
    const float* x    = (const float*)d_in[0];
    const int*   mask = (const int*)  d_in[1];
    const float* ncs  = (const float*)d_in[4];
    const float* ncb  = (const float*)d_in[5];
    const float* dww  = (const float*)d_in[6];
    const float* pww  = (const float*)d_in[7];
    const float* pwb  = (const float*)d_in[8];
    const float* ln1s = (const float*)d_in[9];
    const float* ln1b = (const float*)d_in[10];
    const float* ln2s = (const float*)d_in[11];
    const float* ln2b = (const float*)d_in[12];
    const float* wkv  = (const float*)d_in[13];
    const float* wq   = (const float*)d_in[14];
    const float* f1w  = (const float*)d_in[15];
    const float* f1b  = (const float*)d_in[16];
    const float* f2w  = (const float*)d_in[17];
    const float* f2b  = (const float*)d_in[18];

    char* ws = (char*)d_ws;
    size_t off = 0;
    auto carve = [&](size_t bytes) { char* p = ws + off; off += (bytes + 255) & ~(size_t)255; return p; };
    const size_t EF32 = (size_t)BB * CC * LL * sizeof(float);      // 32 MB
    const size_t EB16 = (size_t)BB * CC * LL * sizeof(short);      // 16 MB
    float*          bufA = (float*)carve(EF32);
    float*          bufB = (float*)carve(EF32);
    float*          bufC = (float*)carve(EF32);
    float*          KVf  = (float*)carve(2 * EF32);
    float*          Qf   = (float*)carve(EF32);
    unsigned short* Xt   = (unsigned short*)carve(EB16);
    unsigned short* Qb   = (unsigned short*)carve(EB16);
    unsigned short* Kb   = (unsigned short*)carve(EB16);
    unsigned short* Vb   = (unsigned short*)carve(EB16);
    unsigned short* Wbf  = (unsigned short*)carve((size_t)2 * CC * CC * sizeof(short));
    (void)ws_size; (void)in_sizes; (void)n_in; (void)out_size;

    const int elems = BB * CC * LL;
    dim3 ln_grid(LL / 256, BB);
    dim3 tr_grid(LL / 32, CC / 32, BB);
    dim3 tr_blk(32, 8);
    int gemmC_blocks  = (BB * (CC / 16)     * (LL / 64)) / 8;   // 1024
    int gemm2C_blocks = (BB * (2 * CC / 16) * (LL / 64)) / 8;   // 2048
    int attn_blocks   = (BB * HH * (LL / 16)) / ATTN_WPB;       // 1024

    // Positional encoding -> bufA (running result)
    k_posenc<<<elems / 256, 256, 0, stream>>>(x, bufA);

    // 4 conv blocks
    for (int i = 0; i < 4; ++i) {
        k_ln_ch<<<ln_grid, 256, 0, stream>>>(bufA, ncs + i * CC, ncb + i * CC, bufB);
        k_dwconv<<<elems / 256, 256, 0, stream>>>(bufB, dww + i * CC * KW, bufC);
        k_transpose<<<tr_grid, tr_blk, 0, stream>>>(bufC, Xt);
        k_f2bf<<<(CC * CC) / 256, 256, 0, stream>>>(pww + i * CC * CC, Wbf, CC * CC);
        k_gemm<<<gemmC_blocks, 256, 0, stream>>>(Wbf, Xt, pwb + i * CC, bufA, bufA, CC, 1);
    }

    // Attention block
    k_ln_ch<<<ln_grid, 256, 0, stream>>>(bufA, ln1s, ln1b, bufB);
    k_transpose<<<tr_grid, tr_blk, 0, stream>>>(bufB, Xt);
    k_f2bf<<<(2 * CC * CC) / 256, 256, 0, stream>>>(wkv, Wbf, 2 * CC * CC);
    k_gemm<<<gemm2C_blocks, 256, 0, stream>>>(Wbf, Xt, nullptr, nullptr, KVf, 2 * CC, 0);
    k_f2bf<<<(CC * CC) / 256, 256, 0, stream>>>(wq, Wbf, CC * CC);
    k_gemm<<<gemmC_blocks, 256, 0, stream>>>(Wbf, Xt, nullptr, nullptr, Qf, CC, 0);
    k_pack<<<elems / 256, 256, 0, stream>>>(Qf, KVf, Qb, Kb, Vb);
    k_attn<<<attn_blocks, 32 * ATTN_WPB, 0, stream>>>(Qb, Kb, Vb, mask, bufA, bufA);

    // FFN block
    k_ln_ch<<<ln_grid, 256, 0, stream>>>(bufA, ln2s, ln2b, bufB);
    k_transpose<<<tr_grid, tr_blk, 0, stream>>>(bufB, Xt);
    k_f2bf<<<(CC * CC) / 256, 256, 0, stream>>>(f1w, Wbf, CC * CC);
    k_gemm<<<gemmC_blocks, 256, 0, stream>>>(Wbf, Xt, f1b, nullptr, bufC, CC, 1);
    k_transpose<<<tr_grid, tr_blk, 0, stream>>>(bufC, Xt);
    k_f2bf<<<(CC * CC) / 256, 256, 0, stream>>>(f2w, Wbf, CC * CC);
    k_gemm<<<gemmC_blocks, 256, 0, stream>>>(Wbf, Xt, f2b, bufA, (float*)d_out, CC, 0);
}
